// SwinTransformerDecoder_26774826124022
// MI455X (gfx1250) — matrix operand order repaired and checked by reading, not compiled, not run on hardware
//
#include <hip/hip_runtime.h>
#include <hip/hip_bf16.h>
#include <math.h>

typedef __attribute__((ext_vector_type(16))) _Float16 v16h;
typedef __attribute__((ext_vector_type(8)))  _Float16 v8h;
typedef __attribute__((ext_vector_type(8)))  float    v8f;

#define NEG_SLOPE 0.01f
#define BN_EPS    1e-5f

// ---------------------------------------------------------------------------
// WMMA GEMM, branch-free double-buffered K loop.
//   A: f16 row-major (M x lda), lda % 32 == 0, M % 16 == 0
//   Bp: f16 packed in WMMA B-fragment order:
//       offset((k,n)) = ((nt*KT + kt)*32 + lane)*16 + h
//       with kt=k/32, kl=k%32, nt=n/16, nl=n%16, lane=nl+16*(kl/16), h=kl%16
//       padded to Npad = ceil(N/64)*64 columns (zeros) -> no predication on loads.
// One wave32 computes a 16x64 tile; stage k+1's 10 b128 loads are issued before
// stage k's 4 WMMAs so the waits only cover the previous stage.
// ---------------------------------------------------------------------------
struct GemmArgs {
  const _Float16* __restrict__ A; int lda;
  const _Float16* __restrict__ Bp;
  const float* __restrict__ bias;   // len N or nullptr
  const float* __restrict__ scale;  // epi==1
  const float* __restrict__ shift;  // epi==1
  _Float16* __restrict__ D; int ldd;
  int M, N;
  int epi;        // 0:+bias  1:+bias->BN->leaky  2:+bias->sigmoid
  int remap;      // 0 id, 1 convT(1,2,2), 2 convT(2,2,2)  (fused upsample scatter)
  int wSh, hSh, dSh, rk, ri, rj;    // input-level dims are powers of two
};

__global__ __launch_bounds__(32) void gemm16x64_wmma(GemmArgs p) {
  const int lane = threadIdx.x;
  const int g = lane >> 4, l = lane & 15;
  const int m0 = (int)blockIdx.y << 4;
  const int n0 = (int)blockIdx.x << 6;
  const int KT = p.lda >> 5;

  // A: lane group g takes k-offsets {16g..16g+15} per 32-k step
  const _Float16* Arow = p.A + (long)(m0 + l) * p.lda + (g << 4);
  // B: per-lane 16 contiguous halves per fragment
  const _Float16* Bbl  = p.Bp + (((long)(n0 >> 4) * KT) << 9) + (lane << 4);

  // stage 0 preload
  v8h  alo = *(const v8h*)(Arow);
  v8h  ahi = *(const v8h*)(Arow + 8);
  v16h bc0 = *(const v16h*)(Bbl + ((long)(0 * KT) << 9));
  v16h bc1 = *(const v16h*)(Bbl + ((long)(1 * KT) << 9));
  v16h bc2 = *(const v16h*)(Bbl + ((long)(2 * KT) << 9));
  v16h bc3 = *(const v16h*)(Bbl + ((long)(3 * KT) << 9));

  v8f acc[4] = {};
  for (int kt = 0; kt < KT; ++kt) {
    const int ktn = (kt + 1 < KT) ? (kt + 1) : kt;   // uniform, branch-free
    // issue ALL next-stage loads before consuming the current stage
    const v8h  nalo = *(const v8h*)(Arow + (ktn << 5));
    const v8h  nahi = *(const v8h*)(Arow + (ktn << 5) + 8);
    const v16h nb0 = *(const v16h*)(Bbl + (((long)(0 * KT + ktn)) << 9));
    const v16h nb1 = *(const v16h*)(Bbl + (((long)(1 * KT + ktn)) << 9));
    const v16h nb2 = *(const v16h*)(Bbl + (((long)(2 * KT + ktn)) << 9));
    const v16h nb3 = *(const v16h*)(Bbl + (((long)(3 * KT + ktn)) << 9));
    __builtin_prefetch(Arow + ((ktn + 1) << 5), 0, 3);

    const v16h a = __builtin_shufflevector(alo, ahi,
        0,1,2,3,4,5,6,7,8,9,10,11,12,13,14,15);
    acc[0] = __builtin_amdgcn_wmma_f32_16x16x32_f16(false, a, false, bc0, (short)0, acc[0], false, false);
    acc[1] = __builtin_amdgcn_wmma_f32_16x16x32_f16(false, a, false, bc1, (short)0, acc[1], false, false);
    acc[2] = __builtin_amdgcn_wmma_f32_16x16x32_f16(false, a, false, bc2, (short)0, acc[2], false, false);
    acc[3] = __builtin_amdgcn_wmma_f32_16x16x32_f16(false, a, false, bc3, (short)0, acc[3], false, false);

    alo = nalo; ahi = nahi; bc0 = nb0; bc1 = nb1; bc2 = nb2; bc3 = nb3;
  }

  // Epilogue + store (C/D layout: VGPR r -> M = r + 8*g, N = l)
#pragma unroll
  for (int t = 0; t < 4; ++t) {
    const int n = n0 + (t << 4) + l;
    if (n >= p.N) continue;
    const float bias = p.bias ? p.bias[n] : 0.f;
    float sc = 1.f, sh = 0.f;
    if (p.epi == 1) { sc = p.scale[n]; sh = p.shift[n]; }
#pragma unroll
    for (int r = 0; r < 8; ++r) {
      const int m = m0 + r + (g << 3);
      float v = acc[t][r] + bias;
      if (p.epi == 1) { v = fmaf(v, sc, sh); v = v > 0.f ? v : NEG_SLOPE * v; }
      else if (p.epi == 2) { v = 1.f / (1.f + __expf(-v)); }
      int row = m;
      if (p.remap == 1) {            // (bd,h,w) -> (bd, 2h+ri, 2w+rj)
        const int w = m & ((1 << p.wSh) - 1); const int q = m >> p.wSh;
        const int hh = q & ((1 << p.hSh) - 1); const int bd = q >> p.hSh;
        row = (((bd << (p.hSh + 1)) + (hh << 1) + p.ri) << (p.wSh + 1))
              + (w << 1) + p.rj;
      } else if (p.remap == 2) {     // (b,d,h,w) -> (b, 2d+rk, 2h+ri, 2w+rj)
        const int w = m & ((1 << p.wSh) - 1); int q = m >> p.wSh;
        const int hh = q & ((1 << p.hSh) - 1); q >>= p.hSh;
        const int d = q & ((1 << p.dSh) - 1); const int bb = q >> p.dSh;
        row = (((((bb << (p.dSh + 1)) + (d << 1) + p.rk) << (p.hSh + 1))
                + (hh << 1) + p.ri) << (p.wSh + 1)) + (w << 1) + p.rj;
      }
      p.D[(long)row * p.ldd + n] = (_Float16)v;
    }
  }
}

// ---------------------------------------------------------------------------
// Weight repack: generalized-stride fp32 weight -> fragment-ordered f16, padded.
// ---------------------------------------------------------------------------
struct RepackArgs {
  const float* __restrict__ W; long bk, bn, boff;
  _Float16* __restrict__ out; int K, N, Kpad, Npad;
};
__global__ void k_repack(RepackArgs p) {
  const long i = (long)blockIdx.x * blockDim.x + threadIdx.x;
  const long total = (long)p.Kpad * p.Npad;
  if (i >= total) return;
  const int k = (int)(i / p.Npad), n = (int)(i % p.Npad);
  const float v = (k < p.K && n < p.N)
      ? p.W[(long)k * p.bk + (long)n * p.bn + p.boff] : 0.f;
  const int kt = k >> 5, kl = k & 31, nt = n >> 4, nl = n & 15;
  const int lane = nl + ((kl >> 4) << 4);
  const int h = kl & 15;
  p.out[((((long)nt * (p.Kpad >> 5) + kt) << 5) + lane) * 16 + h] = (_Float16)v;
}

// ---------------------------------------------------------------------------
// Elementwise helpers (f16 activations)
// ---------------------------------------------------------------------------
__global__ void k_nchw_to_mc(const float* __restrict__ in, _Float16* __restrict__ out,
                             int C, long S, long total) {
  long i = (long)blockIdx.x * blockDim.x + threadIdx.x;
  if (i >= total) return;
  const long s = i % S; const long q = i / S;
  const int  c = (int)(q % C); const long b = q / C;
  out[(b * S + s) * C + c] = (_Float16)in[i];
}

__global__ void k_concat(const _Float16* __restrict__ a, int Ca,
                         const _Float16* __restrict__ b, int Cb,
                         _Float16* __restrict__ out, long total) {
  long i = (long)blockIdx.x * blockDim.x + threadIdx.x;
  if (i >= total) return;
  const int C = Ca + Cb;
  const int c = (int)(i % C); const long m = i / C;
  out[i] = (c < Ca) ? a[m * Ca + c] : b[m * Cb + (c - Ca)];
}

__global__ void k_mix(const _Float16* __restrict__ g, const _Float16* __restrict__ a,
                      const _Float16* __restrict__ b, _Float16* __restrict__ out,
                      long total) {
  long i = (long)blockIdx.x * blockDim.x + threadIdx.x;
  if (i >= total) return;
  const float gv = (float)g[i], av = (float)a[i], bv = (float)b[i];
  out[i] = (_Float16)fmaf(gv, av - bv, bv);      // g*a + (1-g)*b
}

__global__ void k_bnprep(const float* gm, const float* be, const float* rm,
                         const float* rv, float* scale, float* shift, int C) {
  int c = blockIdx.x * blockDim.x + threadIdx.x;
  if (c >= C) return;
  const float s = gm[c] * rsqrtf(rv[c] + BN_EPS);
  scale[c] = s; shift[c] = be[c] - rm[c] * s;
}

__global__ void k_zero_h(_Float16* __restrict__ p, long n) {
  long i = (long)blockIdx.x * blockDim.x + threadIdx.x;
  if (i < n) p[i] = (_Float16)0.f;
}

__global__ void k_final_sigmoid(const _Float16* __restrict__ x, const float* w2,
                                const float* b2, float* __restrict__ out, long total) {
  long i = (long)blockIdx.x * blockDim.x + threadIdx.x;
  if (i >= total) return;
  const float v = w2[0] * (float)x[i] + b2[0];
  out[i] = 1.f / (1.f + __expf(-v));
}

// ---------------------------------------------------------------------------
// Host-side orchestration
// ---------------------------------------------------------------------------
static inline void gemm(hipStream_t s, const _Float16* A, int lda, const _Float16* Bp,
                        const float* bias, const float* scale, const float* shift,
                        _Float16* D, int ldd, int M, int N, int epi,
                        int remap = 0, int wSh = 0, int hSh = 0, int dSh = 0,
                        int rk = 0, int ri = 0, int rj = 0) {
  GemmArgs p{A, lda, Bp, bias, scale, shift, D, ldd, M, N, epi,
             remap, wSh, hSh, dSh, rk, ri, rj};
  dim3 grid((N + 63) / 64, M >> 4);
  gemm16x64_wmma<<<grid, dim3(32), 0, s>>>(p);
}

static inline void repack(hipStream_t s, const float* W, long bk, long bn, long boff,
                          _Float16* out, int K, int N) {
  const int Kp = ((K + 31) / 32) * 32, Np = ((N + 63) / 64) * 64;
  RepackArgs p{W, bk, bn, boff, out, K, N, Kp, Np};
  const long total = (long)Kp * Np;
  k_repack<<<dim3((unsigned)((total + 255) / 256)), 256, 0, s>>>(p);
}

static inline void ew_grid(long n, dim3& g, dim3& b) {
  b = dim3(256); g = dim3((unsigned)((n + 255) / 256));
}

// input index map (setup_inputs dict flatten order)
enum {
  IN_FEATURE = 0, IN_X1, IN_X2, IN_X3,
  SC1_W, SC1_B, SC1_G, SC1_BE, SC1_RM, SC1_RV,
  SC2_W, SC2_B, SC2_G, SC2_BE, SC2_RM, SC2_RV,
  SC3_W, SC3_B, SC3_G, SC3_BE, SC3_RM, SC3_RV,
  G2_W0, G2_W1, G2_W2, G2_B0, G2_B1, G2_B2, G2_FW, G2_FB,
  G1_W0, G1_W1, G1_W2, G1_B0, G1_B1, G1_B2, G1_FW, G1_FB,
  G3_W0, G3_W1, G3_W2, G3_B0, G3_B1, G3_B2, G3_FW, G3_FB,
  DB1_W1, DB1_B1, DB1_WT, DB1_BT, DB1_W2, DB1_B2, DB1_G, DB1_BE, DB1_RM, DB1_RV,
  DB2_W1, DB2_B1, DB2_WT, DB2_BT, DB2_W2, DB2_B2, DB2_G, DB2_BE, DB2_RM, DB2_RV,
  DB3_W1, DB3_B1, DB3_WT, DB3_BT, DB3_W2, DB3_B2
};

extern "C" void kernel_launch(void* const* d_in, const int* in_sizes, int n_in,
                              void* d_out, int out_size, void* d_ws, size_t ws_size,
                              hipStream_t stream) {
  auto F = [&](int i) { return (const float*)d_in[i]; };

  const int Bb = 2, Dd = 8, H3 = 16, W3 = 16;
  const long S3 = (long)Dd * H3 * W3;          // 2048
  const long S2 = (long)Dd * 32 * 32;          // 8192
  const long S1 = (long)Dd * 64 * 64;          // 32768
  const int  M3 = (int)(Bb * S3);              // 4096
  const int  M2 = (int)(Bb * S2);              // 16384
  const int  M1 = (int)(Bb * S1);              // 65536
  const long M0 = (long)Bb * 16 * 128 * 128;   // 524288

  // bump allocator (256-byte aligned) over workspace
  char* base = (char*)d_ws;
  size_t off = 0;
  auto allocB = [&](size_t bytes) {
    off = (off + 255) & ~(size_t)255;
    void* p = base + off; off += bytes; return p;
  };
  auto allocH = [&](long n) { return (_Float16*)allocB((size_t)n * 2); };
  auto allocF = [&](long n) { return (float*)allocB((size_t)n * 4); };
  auto psz = [&](int K, int N) {
    return (long)(((K + 31) / 32) * 32) * (((N + 63) / 64) * 64);
  };
  (void)ws_size; (void)n_in; (void)in_sizes; (void)out_size;

  // activations (f16, channels-last)
  _Float16* featT = allocH((long)M3 * 384);
  _Float16* x3T   = allocH((long)M3 * 384);
  _Float16* x2T   = allocH((long)M2 * 192);
  _Float16* x1T   = allocH((long)M1 * 96);
  _Float16* sc3o  = allocH((long)M3 * 384);
  _Float16* sc2o  = allocH((long)M2 * 192);
  _Float16* sc1o  = allocH((long)M1 * 96);
  _Float16* cat2  = allocH((long)M3 * 768);
  _Float16* hd2   = allocH((long)M3 * 1152);
  _Float16* gt2   = allocH((long)M3 * 384);
  _Float16* fA    = allocH((long)M3 * 384);
  _Float16* t1    = allocH((long)M3 * 192);
  _Float16* f2    = allocH((long)M2 * 192);
  _Float16* u2    = allocH((long)M2 * 192);
  _Float16* cat1  = allocH((long)M2 * 384);
  _Float16* hd1   = allocH((long)M2 * 576);
  _Float16* gt1   = allocH((long)M2 * 192);
  _Float16* fB    = allocH((long)M2 * 192);
  _Float16* t2    = allocH((long)M2 * 96);
  _Float16* f1    = allocH((long)M1 * 96);
  _Float16* u1    = allocH((long)M1 * 96);
  _Float16* cat3  = allocH((long)M1 * 192);
  _Float16* hd3   = allocH((long)M1 * 288);
  _Float16* gt3   = allocH((long)M1 * 96);
  _Float16* fC    = allocH((long)M1 * 96);
  _Float16* t3    = allocH((long)M1 * 64);     // K padded 48->64
  _Float16* pre   = allocH(M0);
  // packed weights (f16, fragment order)
  _Float16* Psc1 = allocH(psz(96, 96));
  _Float16* Psc2 = allocH(psz(192, 192));
  _Float16* Psc3 = allocH(psz(384, 384));
  _Float16* Pg2w[3] = {allocH(psz(768, 384)), allocH(psz(768, 384)), allocH(psz(768, 384))};
  _Float16* Pg2f = allocH(psz(1152, 384));
  _Float16* Pg1w[3] = {allocH(psz(384, 192)), allocH(psz(384, 192)), allocH(psz(384, 192))};
  _Float16* Pg1f = allocH(psz(576, 192));
  _Float16* Pg3w[3] = {allocH(psz(192, 96)), allocH(psz(192, 96)), allocH(psz(192, 96))};
  _Float16* Pg3f = allocH(psz(288, 96));
  _Float16* Pd1w1 = allocH(psz(384, 192));
  _Float16* Pd1wt[4] = {allocH(psz(192, 192)), allocH(psz(192, 192)),
                        allocH(psz(192, 192)), allocH(psz(192, 192))};
  _Float16* Pd1w2 = allocH(psz(192, 192));
  _Float16* Pd2w1 = allocH(psz(192, 96));
  _Float16* Pd2wt[4] = {allocH(psz(96, 96)), allocH(psz(96, 96)),
                        allocH(psz(96, 96)), allocH(psz(96, 96))};
  _Float16* Pd2w2 = allocH(psz(96, 96));
  _Float16* Pd3w1 = allocH(psz(96, 48));
  _Float16* Pd3wt[8];
  for (int s = 0; s < 8; ++s) Pd3wt[s] = allocH(psz(48, 1));
  // BN scale/shift
  float* sc1S = allocF(96),  * sc1H = allocF(96);
  float* sc2S = allocF(192), * sc2H = allocF(192);
  float* sc3S = allocF(384), * sc3H = allocF(384);
  float* db1S = allocF(192), * db1H = allocF(192);
  float* db2S = allocF(96),  * db2H = allocF(96);

  dim3 g, b;

  // --- weight repacks (conv1x1 W is (O,C) -> B[k][n]=W[n][k]: bk=1, bn=C) ---
  repack(stream, F(SC1_W), 1, 96,  0, Psc1, 96, 96);
  repack(stream, F(SC2_W), 1, 192, 0, Psc2, 192, 192);
  repack(stream, F(SC3_W), 1, 384, 0, Psc3, 384, 384);
  const int G2W[3] = {G2_W0, G2_W1, G2_W2}, G1W[3] = {G1_W0, G1_W1, G1_W2},
            G3W[3] = {G3_W0, G3_W1, G3_W2};
  for (int i = 0; i < 3; ++i) {
    repack(stream, F(G2W[i]), 384, 1, 0, Pg2w[i], 768, 384);   // gate W is (cin,h)
    repack(stream, F(G1W[i]), 192, 1, 0, Pg1w[i], 384, 192);
    repack(stream, F(G3W[i]), 96,  1, 0, Pg3w[i], 192, 96);
  }
  repack(stream, F(G2_FW), 384, 1, 0, Pg2f, 1152, 384);
  repack(stream, F(G1_FW), 192, 1, 0, Pg1f, 576, 192);
  repack(stream, F(G3_FW), 96,  1, 0, Pg3f, 288, 96);
  repack(stream, F(DB1_W1), 1, 384, 0, Pd1w1, 384, 192);
  repack(stream, F(DB1_W2), 1, 192, 0, Pd1w2, 192, 192);
  repack(stream, F(DB2_W1), 1, 192, 0, Pd2w1, 192, 96);
  repack(stream, F(DB2_W2), 1, 96,  0, Pd2w2, 96, 96);
  repack(stream, F(DB3_W1), 1, 96,  0, Pd3w1, 96, 48);
  for (int s = 0; s < 4; ++s) {                  // wt (C,O,2,2): (k,n)->(k*O+n)*4+s
    repack(stream, F(DB1_WT), 192 * 4, 4, s, Pd1wt[s], 192, 192);
    repack(stream, F(DB2_WT), 96 * 4,  4, s, Pd2wt[s], 96, 96);
  }
  for (int s = 0; s < 8; ++s)                    // wt (48,1,2,2,2): k*8+s
    repack(stream, F(DB3_WT), 8, 8, s, Pd3wt[s], 48, 1);

  // --- BN scale/shift precompute ---
  k_bnprep<<<1, 128, 0, stream>>>(F(SC1_G), F(SC1_BE), F(SC1_RM), F(SC1_RV), sc1S, sc1H, 96);
  k_bnprep<<<1, 256, 0, stream>>>(F(SC2_G), F(SC2_BE), F(SC2_RM), F(SC2_RV), sc2S, sc2H, 192);
  k_bnprep<<<2, 256, 0, stream>>>(F(SC3_G), F(SC3_BE), F(SC3_RM), F(SC3_RV), sc3S, sc3H, 384);
  k_bnprep<<<1, 256, 0, stream>>>(F(DB1_G), F(DB1_BE), F(DB1_RM), F(DB1_RV), db1S, db1H, 192);
  k_bnprep<<<1, 128, 0, stream>>>(F(DB2_G), F(DB2_BE), F(DB2_RM), F(DB2_RV), db2S, db2H, 96);

  // --- NCHW fp32 -> (M,C) f16 ---
  ew_grid((long)Bb * 384 * S3, g, b); k_nchw_to_mc<<<g, b, 0, stream>>>(F(IN_FEATURE), featT, 384, S3, (long)Bb * 384 * S3);
  ew_grid((long)Bb * 384 * S3, g, b); k_nchw_to_mc<<<g, b, 0, stream>>>(F(IN_X3),      x3T,   384, S3, (long)Bb * 384 * S3);
  ew_grid((long)Bb * 192 * S2, g, b); k_nchw_to_mc<<<g, b, 0, stream>>>(F(IN_X2),      x2T,   192, S2, (long)Bb * 192 * S2);
  ew_grid((long)Bb * 96  * S1, g, b); k_nchw_to_mc<<<g, b, 0, stream>>>(F(IN_X1),      x1T,   96,  S1, (long)Bb * 96  * S1);

  // --- skip connections: conv1x1 + BN + LeakyReLU ---
  gemm(stream, x1T, 96,  Psc1, F(SC1_B), sc1S, sc1H, sc1o, 96,  M1, 96,  1);
  gemm(stream, x2T, 192, Psc2, F(SC2_B), sc2S, sc2H, sc2o, 192, M2, 192, 1);
  gemm(stream, x3T, 384, Psc3, F(SC3_B), sc3S, sc3H, sc3o, 384, M3, 384, 1);

  // ---- Stage 1 (level 3): gate g2 + db1 ----
  ew_grid((long)M3 * 768, g, b); k_concat<<<g, b, 0, stream>>>(featT, 384, sc3o, 384, cat2, (long)M3 * 768);
  gemm(stream, cat2, 768, Pg2w[0], F(G2_B0), nullptr, nullptr, hd2 + 0,   1152, M3, 384, 2);
  gemm(stream, cat2, 768, Pg2w[1], F(G2_B1), nullptr, nullptr, hd2 + 384, 1152, M3, 384, 2);
  gemm(stream, cat2, 768, Pg2w[2], F(G2_B2), nullptr, nullptr, hd2 + 768, 1152, M3, 384, 2);
  gemm(stream, hd2, 1152, Pg2f, F(G2_FB), nullptr, nullptr, gt2, 384, M3, 384, 0);
  ew_grid((long)M3 * 384, g, b); k_mix<<<g, b, 0, stream>>>(gt2, featT, sc3o, fA, (long)M3 * 384);
  gemm(stream, fA, 384, Pd1w1, F(DB1_B1), nullptr, nullptr, t1, 192, M3, 192, 0);
  for (int s = 0; s < 4; ++s)
    gemm(stream, t1, 192, Pd1wt[s], F(DB1_BT), nullptr, nullptr,
         f2, 192, M3, 192, 0, /*remap*/1, /*wSh*/4, /*hSh*/4, 0, 0, s >> 1, s & 1);
  gemm(stream, f2, 192, Pd1w2, F(DB1_B2), db1S, db1H, u2, 192, M2, 192, 1);

  // ---- Stage 2 (level 2): gate g1 + db2 ----
  ew_grid((long)M2 * 384, g, b); k_concat<<<g, b, 0, stream>>>(u2, 192, sc2o, 192, cat1, (long)M2 * 384);
  gemm(stream, cat1, 384, Pg1w[0], F(G1_B0), nullptr, nullptr, hd1 + 0,   576, M2, 192, 2);
  gemm(stream, cat1, 384, Pg1w[1], F(G1_B1), nullptr, nullptr, hd1 + 192, 576, M2, 192, 2);
  gemm(stream, cat1, 384, Pg1w[2], F(G1_B2), nullptr, nullptr, hd1 + 384, 576, M2, 192, 2);
  gemm(stream, hd1, 576, Pg1f, F(G1_FB), nullptr, nullptr, gt1, 192, M2, 192, 0);
  ew_grid((long)M2 * 192, g, b); k_mix<<<g, b, 0, stream>>>(gt1, u2, sc2o, fB, (long)M2 * 192);
  gemm(stream, fB, 192, Pd2w1, F(DB2_B1), nullptr, nullptr, t2, 96, M2, 96, 0);
  for (int s = 0; s < 4; ++s)
    gemm(stream, t2, 96, Pd2wt[s], F(DB2_BT), nullptr, nullptr,
         f1, 96, M2, 96, 0, /*remap*/1, /*wSh*/5, /*hSh*/5, 0, 0, s >> 1, s & 1);
  gemm(stream, f1, 96, Pd2w2, F(DB2_B2), db2S, db2H, u1, 96, M1, 96, 1);

  // ---- Stage 3 (level 1): gate g3 + db3 ----
  ew_grid((long)M1 * 192, g, b); k_concat<<<g, b, 0, stream>>>(u1, 96, sc1o, 96, cat3, (long)M1 * 192);
  gemm(stream, cat3, 192, Pg3w[0], F(G3_B0), nullptr, nullptr, hd3 + 0,   288, M1, 96, 2);
  gemm(stream, cat3, 192, Pg3w[1], F(G3_B1), nullptr, nullptr, hd3 + 96,  288, M1, 96, 2);
  gemm(stream, cat3, 192, Pg3w[2], F(G3_B2), nullptr, nullptr, hd3 + 192, 288, M1, 96, 2);
  gemm(stream, hd3, 288, Pg3f, F(G3_FB), nullptr, nullptr, gt3, 96, M1, 96, 0);
  ew_grid((long)M1 * 96, g, b); k_mix<<<g, b, 0, stream>>>(gt3, u1, sc1o, fC, (long)M1 * 96);
  // t3 has K padded 48->64: zero-fill so padded columns contribute 0
  ew_grid((long)M1 * 64, g, b); k_zero_h<<<g, b, 0, stream>>>(t3, (long)M1 * 64);
  gemm(stream, fC, 96, Pd3w1, F(DB3_B1), nullptr, nullptr, t3, 64, M1, 48, 0);
  for (int s = 0; s < 8; ++s)
    gemm(stream, t3, 64, Pd3wt[s], F(DB3_BT), nullptr, nullptr,
         pre, 1, M1, 1, 0, /*remap*/2, /*wSh*/6, /*hSh*/6, /*dSh*/3,
         (s >> 2) & 1, (s >> 1) & 1, s & 1);
  // final scalar 1x1 conv + sigmoid -> d_out (2,1,16,128,128)
  ew_grid(M0, g, b);
  k_final_sigmoid<<<g, b, 0, stream>>>(pre, F(DB3_W2), F(DB3_B2), (float*)d_out, M0);
}
